// _INT8LSTMCell_73194832659130
// MI455X (gfx1250) — compile-verified, matrix operand order checked by reading, not run on hardware
//
#include <hip/hip_runtime.h>
#include <cstdint>

typedef __attribute__((ext_vector_type(16))) _Float16 v16h;
typedef __attribute__((ext_vector_type(8)))  _Float16 v8h;
typedef __attribute__((ext_vector_type(4)))  _Float16 v4h;
typedef __attribute__((ext_vector_type(8)))  float    v8f;

constexpr int BATCH  = 16384;
constexpr int HIDDEN = 512;
constexpr int INPUT  = 512;
constexpr int KTOT   = INPUT + HIDDEN;  // 1024 (x | h_prev concatenated along K)
constexpr int BM     = 128;             // batch rows per workgroup
constexpr int BN     = 64;              // hidden cols per workgroup (x4 gates)
constexpr int BK     = 64;              // K per pipeline stage
constexpr int LDSS   = 72;              // f16 elems per LDS row (64 data + 8 pad)

#if __has_builtin(__builtin_amdgcn_sched_group_barrier)
#define SCHED_GROUP(mask, size, id) __builtin_amdgcn_sched_group_barrier(mask, size, id)
#else
#define SCHED_GROUP(mask, size, id)
#endif
#define SG_MFMA    0x008  // MFMA/WMMA category
#define SG_DSREAD  0x100  // DS read category

__device__ __forceinline__ float sigmoid_f(float x) {
  return 1.0f / (1.0f + __expf(-x));
}
__device__ __forceinline__ float tanh_f(float x) {
  float a = fabsf(x);
  float e = __expf(-2.0f * a);
  float t = (1.0f - e) / (1.0f + e);
  return copysignf(t, x);
}

// Load one 16x32 f16 WMMA operand fragment from an LDS tile (row-major,
// LDSS stride). CDNA5 16-bit A/B layout: lane<16 holds row r, K=[ks..ks+7]
// and [ks+16..ks+23]; lane>=16 holds K=[ks+8..ks+15] and [ks+24..ks+31].
__device__ __forceinline__ v16h load_frag(const _Float16* tile, int rowBase,
                                          int r, int half, int ks) {
  const _Float16* p = tile + (size_t)(rowBase + r) * LDSS + ks + half * 8;
  v8h lo = *(const v8h*)p;          // 16B ds_load_b128
  v8h hi = *(const v8h*)(p + 16);   // 16B ds_load_b128
  return __builtin_shufflevector(lo, hi, 0, 1, 2, 3, 4, 5, 6, 7,
                                         8, 9, 10, 11, 12, 13, 14, 15);
}

__global__ __launch_bounds__(256) void lstm_wmma_kernel(
    const float* __restrict__ x, const float* __restrict__ h_prev,
    const float* __restrict__ c_prev, const float* __restrict__ w_ih,
    const float* __restrict__ w_hh, const float* __restrict__ b_ih,
    const float* __restrict__ b_hh, float* __restrict__ h_out,
    float* __restrict__ c_out) {
  // Double-buffered tiles: 2 x (18 KB + 36 KB) = 108 KB of the WGP's 320 KB.
  __shared__ _Float16 ldsA[2][BM * LDSS];
  __shared__ _Float16 ldsB[2][4 * BN * LDSS];

  const int tid  = threadIdx.x;
  const int lane = tid & 31;
  const int wave = tid >> 5;
  const int wm   = wave & 1;   // 2 M-subtiles of 64 rows
  const int wn   = wave >> 1;  // 4 hidden subtiles of 16 cols
  const int r    = lane & 15;
  const int half = lane >> 4;

  const int mBase   = blockIdx.y * BM;   // batch-row base
  const int hidBase = blockIdx.x * BN;   // hidden-col base
  const int j       = hidBase + wn * 16 + r;  // this lane's hidden column

  // Accumulators [gate][mi]; C/D layout: N = lane&15, M = half*8 + e.
  // Bias depends only on N -> fold (bias_ih + bias_hh) into the init.
  v8f acc[4][4];
#pragma unroll
  for (int g = 0; g < 4; ++g) {
    const float tb = b_ih[g * HIDDEN + j] + b_hh[g * HIDDEN + j];
    v8f init = {tb, tb, tb, tb, tb, tb, tb, tb};
#pragma unroll
    for (int mi = 0; mi < 4; ++mi) acc[g][mi] = init;
  }

  // Stage fill: global fp32 -> registers -> f16 -> LDS.
  auto fill = [&](int kk, _Float16* lA, _Float16* lB) {
    const bool first   = kk < INPUT;
    const float* baseA = first ? x : h_prev;
    const float* baseB = first ? w_ih : w_hh;
    const int kcol     = first ? kk : kk - INPUT;
    // A: 128 rows x 16 float4
#pragma unroll
    for (int i = 0; i < 8; ++i) {
      int gid = tid + i * 256;
      int row = gid >> 4, c4 = gid & 15;
      float4 f = *(const float4*)(baseA + (size_t)(mBase + row) * INPUT +
                                  kcol + c4 * 4);
      v4h hq = {(_Float16)f.x, (_Float16)f.y, (_Float16)f.z, (_Float16)f.w};
      *(v4h*)&lA[row * LDSS + c4 * 4] = hq;
    }
    // B: 256 rows (4 gates x 64 hidden cols) x 16 float4.
    // gates = act . W^T: column n of B == row n of W (K-contiguous), so B
    // fragments use the same row-major addressing as A.
#pragma unroll
    for (int i = 0; i < 16; ++i) {
      int gid  = tid + i * 256;
      int grow = gid >> 4, c4 = gid & 15;
      int g = grow >> 6, c = grow & 63;
      float4 f = *(const float4*)(baseB +
                                  (size_t)(g * HIDDEN + hidBase + c) * INPUT +
                                  kcol + c4 * 4);
      v4h hq = {(_Float16)f.x, (_Float16)f.y, (_Float16)f.z, (_Float16)f.w};
      *(v4h*)&lB[grow * LDSS + c4 * 4] = hq;
    }
    // Pull the stage after next toward L2 (global_prefetch_b8).
    if (kk + BK < KTOT) {
      const bool nf   = (kk + BK) < INPUT;
      const float* nA = nf ? x : h_prev;
      const float* nB = nf ? w_ih : w_hh;
      const int nk    = nf ? (kk + BK) : (kk + BK - INPUT);
      __builtin_prefetch(nA + (size_t)(mBase + (tid >> 1)) * INPUT + nk +
                             (tid & 1) * 32, 0, 3);
      __builtin_prefetch(nB + (size_t)((tid >> 6) * HIDDEN + hidBase +
                                       (tid & 63)) * INPUT + nk, 0, 3);
    }
  };

  // Stage compute. Per 32-K chunk: 4 A-frags + 4 B-frags (16 ds_load_b128),
  // then 16 WMMAs; each B fragment feeds 4 consecutive WMMAs. Scheduling
  // groups pin "all DS reads, then all WMMAs" so the fragments stay in
  // distinct registers and WMMAs retire on staged dscnt waits instead of a
  // full drain per load/use pair.
  auto compute = [&](const _Float16* lA, const _Float16* lB) {
#pragma unroll
    for (int ks = 0; ks < BK; ks += 32) {
      v16h af[4];
#pragma unroll
      for (int mi = 0; mi < 4; ++mi)
        af[mi] = load_frag(lA, wm * 64 + mi * 16, r, half, ks);
      v16h bf[4];
#pragma unroll
      for (int g = 0; g < 4; ++g)
        bf[g] = load_frag(lB, g * BN + wn * 16, r, half, ks);
#pragma unroll
      for (int g = 0; g < 4; ++g)
#pragma unroll
        for (int mi = 0; mi < 4; ++mi)
          acc[g][mi] = __builtin_amdgcn_wmma_f32_16x16x32_f16(
              false, af[mi], false, bf[g], (short)0, acc[g][mi], false, false);
      SCHED_GROUP(SG_DSREAD, 16, 0);  // the chunk's 16 ds_load_b128 first
      SCHED_GROUP(SG_MFMA,   16, 0);  // then its 16 WMMAs
    }
  };

  // Software pipeline, one barrier per stage:
  //   fill(b1,k+64); compute(b0,k); sync; fill(b0,k+128); compute(b1,k+64); sync
  // Every fill of a buffer is separated from all waves' previous compute of
  // that same buffer by exactly one intervening barrier.
  fill(0, ldsA[0], ldsB[0]);
  __syncthreads();
#pragma unroll 1
  for (int kk = 0; kk < KTOT; kk += 2 * BK) {
    fill(kk + BK, ldsA[1], ldsB[1]);   // next stage's loads issue first
    compute(ldsA[0], ldsB[0]);
    __syncthreads();
    if (kk + 2 * BK < KTOT) fill(kk + 2 * BK, ldsA[0], ldsB[0]);
    compute(ldsA[1], ldsB[1]);
    __syncthreads();
  }

  // Fused LSTM epilogue: all four gates resident in registers.
#pragma unroll
  for (int mi = 0; mi < 4; ++mi) {
#pragma unroll
    for (int e = 0; e < 8; ++e) {
      const int m = mBase + wm * 64 + mi * 16 + half * 8 + e;
      const size_t idx = (size_t)m * HIDDEN + j;
      float ig = sigmoid_f(acc[0][mi][e]);
      float fg = sigmoid_f(acc[1][mi][e]);
      float gg = tanh_f(acc[2][mi][e]);
      float og = sigmoid_f(acc[3][mi][e]);
      float cn = fg * c_prev[idx] + ig * gg;
      h_out[idx] = og * tanh_f(cn);
      c_out[idx] = cn;
    }
  }
}

extern "C" void kernel_launch(void* const* d_in, const int* in_sizes, int n_in,
                              void* d_out, int out_size, void* d_ws,
                              size_t ws_size, hipStream_t stream) {
  (void)in_sizes; (void)n_in; (void)out_size; (void)d_ws; (void)ws_size;
  const float* x      = (const float*)d_in[0];
  const float* h_prev = (const float*)d_in[1];
  const float* c_prev = (const float*)d_in[2];
  const float* w_ih   = (const float*)d_in[3];
  const float* w_hh   = (const float*)d_in[4];
  const float* b_ih   = (const float*)d_in[5];
  const float* b_hh   = (const float*)d_in[6];
  float* h_out = (float*)d_out;
  float* c_out = h_out + (size_t)BATCH * HIDDEN;

  dim3 grid(HIDDEN / BN, BATCH / BM);  // (8, 128) = 1024 workgroups
  lstm_wmma_kernel<<<grid, dim3(256), 0, stream>>>(
      x, h_prev, c_prev, w_ih, w_hh, b_ih, b_hh, h_out, c_out);
}
